// TransformerEncoderDecoder_80066780332704
// MI455X (gfx1250) — matrix operand order repaired and checked by reading, not compiled, not run on hardware
//
#include <hip/hip_runtime.h>

typedef __attribute__((ext_vector_type(16))) _Float16 v16h;
typedef __attribute__((ext_vector_type(8)))  _Float16 h8;
typedef __attribute__((ext_vector_type(8)))  float    v8f;

enum { ACT_NONE = 0, ACT_RELU = 1, ACT_GELU = 2, ACT_ELU1 = 3 };

// ---------------------------------------------------------------------------
// Generic batched GEMM on f16 inputs, f32 accumulate:
//   acc = A @ Bt           A: f16 [M,K] (row stride a_rs)
//                          Bt: f16 [K,N] (pre-transposed weights, row stride N)
//   val = act(acc * rowscale + bias) + residual
//   stores: f32 out32 and/or f16 out16 (same [M,N] layout, row stride out_rs)
// One wave computes a 32x32 tile: 4x v_wmma_f32_16x16x32_f16 per k-step.
// Fragment layouts per CDNA5 ISA:
//   A: lane m = lane&15 (+16 for second tile); elements 0..7 = K akb+0..7,
//      elements 8..15 = K 16+akb+0..7, akb = (lane>>4)*8  -> two 16B loads
//   B: lane n = lane&15 (+16 for b1); elements = same K as A -> strided loads
//   C/D: VGPR r -> row r (+8 for lanes>=16), lane&15 -> column
template <int ACT, bool BIAS, bool SCALE, bool RES, bool O32, bool O16>
__global__ __launch_bounds__(128) void gemm_wmma_kernel(
    float* out32, _Float16* out16,
    const _Float16* __restrict__ A, const _Float16* __restrict__ Bt,
    const float* __restrict__ bias, const float* __restrict__ rowscale,
    const float* residual,
    int M, int N, int K, int a_rs, int out_rs,
    long a_bo, long a_bi, long b_bo, long b_bi,
    long o_bo, long o_bi, long sc_bo, long sc_bi, int sc_rs,
    int batch_inner)
{
  const int lane = threadIdx.x & 31;
  const int wid  = threadIdx.x >> 5;
  const int tn_cnt = N >> 5;
  const int tm_cnt = M >> 5;
  const int tile = blockIdx.x * 4 + wid;
  if (tile >= tm_cnt * tn_cnt) return;
  const int tm = (tile / tn_cnt) << 5;
  const int tn = (tile % tn_cnt) << 5;

  const int zo = blockIdx.z / batch_inner;
  const int zi = blockIdx.z % batch_inner;

  const int mlo = lane & 15;
  const int akb = (lane >> 4) << 3;          // 0 or 8
  const _Float16* ar0 = A + zo * a_bo + zi * a_bi + (long)(tm + mlo) * a_rs;
  const _Float16* ar1 = ar0 + (long)16 * a_rs;
  const _Float16* bp  = Bt + zo * b_bo + zi * b_bi;

  v8f acc00 = {}, acc01 = {}, acc10 = {}, acc11 = {};

  for (int k0 = 0; k0 < K; k0 += 32) {
    const h8 a0lo = *(const h8*)(ar0 + k0 + akb);
    const h8 a0hi = *(const h8*)(ar0 + k0 + 16 + akb);
    const h8 a1lo = *(const h8*)(ar1 + k0 + akb);
    const h8 a1hi = *(const h8*)(ar1 + k0 + 16 + akb);
    const _Float16* bcol = bp + (long)k0 * N + tn + mlo;
    v16h b0, b1;
#pragma unroll
    for (int s = 0; s < 16; ++s) {
      const int kk = (s < 8) ? (akb + s) : (16 + akb + (s - 8));
      b0[s] = bcol[(long)kk * N];
      b1[s] = bcol[(long)kk * N + 16];
    }
    const v16h a0 = __builtin_shufflevector(a0lo, a0hi, 0,1,2,3,4,5,6,7,8,9,10,11,12,13,14,15);
    const v16h a1 = __builtin_shufflevector(a1lo, a1hi, 0,1,2,3,4,5,6,7,8,9,10,11,12,13,14,15);
    acc00 = __builtin_amdgcn_wmma_f32_16x16x32_f16(false, a0, false, b0, (short)0, acc00, false, false);
    acc01 = __builtin_amdgcn_wmma_f32_16x16x32_f16(false, a0, false, b1, (short)0, acc01, false, false);
    acc10 = __builtin_amdgcn_wmma_f32_16x16x32_f16(false, a1, false, b0, (short)0, acc10, false, false);
    acc11 = __builtin_amdgcn_wmma_f32_16x16x32_f16(false, a1, false, b1, (short)0, acc11, false, false);
  }

  const int em_base = tm + ((lane >> 4) << 3);
  const int en_base = tn + mlo;
  float* op32 = nullptr;
  _Float16* op16 = nullptr;
  const float* rp = nullptr;
  const float* scp = nullptr;
  if constexpr (O32)   op32 = out32 + zo * o_bo + zi * o_bi;
  if constexpr (O16)   op16 = out16 + zo * o_bo + zi * o_bi;
  if constexpr (RES)   rp   = residual + zo * o_bo + zi * o_bi;
  if constexpr (SCALE) scp  = rowscale + zo * sc_bo + zi * sc_bi;

#pragma unroll
  for (int ti = 0; ti < 2; ++ti) {
#pragma unroll
    for (int tj = 0; tj < 2; ++tj) {
      const v8f a = ti ? (tj ? acc11 : acc10) : (tj ? acc01 : acc00);
      const int en = en_base + tj * 16;
      float bval = 0.f;
      if constexpr (BIAS) bval = bias[en];
#pragma unroll
      for (int r = 0; r < 8; ++r) {
        const int em = em_base + ti * 16 + r;
        float val = a[r];
        if constexpr (SCALE) val *= scp[(long)em * sc_rs];
        if constexpr (BIAS)  val += bval;
        if constexpr (ACT == ACT_RELU) val = fmaxf(val, 0.f);
        else if constexpr (ACT == ACT_GELU) val = 0.5f * val * (1.f + erff(val * 0.70710678118654752f));
        else if constexpr (ACT == ACT_ELU1) val = val > 0.f ? val + 1.f : __expf(val);
        const long oidx = (long)em * out_rs + en;
        if constexpr (RES) val += rp[oidx];
        if constexpr (O32) op32[oidx] = val;
        if constexpr (O16) op16[oidx] = (_Float16)val;
      }
    }
  }
}

// ---------------------------------------------------------------------------
// Weight f32[N,K] -> f16[K,N] transpose-convert (LDS 32x32 tile).
__global__ __launch_bounds__(256) void w_transpose_f16_kernel(
    const float* __restrict__ src, _Float16* __restrict__ dst, int Nn, int Kk)
{
  __shared__ float tile[32][33];
  const int tx = threadIdx.x & 31, ty = threadIdx.x >> 5;
  const int k0 = blockIdx.x * 32, n0 = blockIdx.y * 32;
#pragma unroll
  for (int i = ty; i < 32; i += 8)
    tile[i][tx] = src[(long)(n0 + i) * Kk + k0 + tx];
  __syncthreads();
#pragma unroll
  for (int i = ty; i < 32; i += 8)
    dst[(long)(k0 + i) * Nn + n0 + tx] = (_Float16)tile[tx][i];
}

// Elementwise f32 -> f16 (same layout).
__global__ void cvt_f16_kernel(const float* __restrict__ s, _Float16* __restrict__ d, long n)
{
  const long i = (long)blockIdx.x * blockDim.x + threadIdx.x;
  if (i < n) d[i] = (_Float16)s[i];
}

// ---------------------------------------------------------------------------
// Non-causal attention: KVT(f16,[d][m]) = sum_s K[s,d]*V[s,m]; Ksum[d](f32).
// One block per (b,h); K/V f16 [B,L,8,64] flattened ([BL,512]).
__global__ __launch_bounds__(256) void kvt_kernel(
    const _Float16* __restrict__ k, const _Float16* __restrict__ v,
    _Float16* __restrict__ kvt, float* __restrict__ ksum, int L)
{
  const int bh = blockIdx.x;
  const int b = bh >> 3, h = bh & 7;
  const _Float16* kp = k + (long)b * L * 512 + h * 64;
  const _Float16* vp = v + (long)b * L * 512 + h * 64;
  const int m  = threadIdx.x >> 2;          // 0..63
  const int d0 = (threadIdx.x & 3) << 4;    // 0,16,32,48
  float acc[16], ks[16];
#pragma unroll
  for (int j = 0; j < 16; ++j) { acc[j] = 0.f; ks[j] = 0.f; }
  const bool doks = (m == 0);
  for (int s = 0; s < L; ++s) {
    const _Float16* kr = kp + (long)s * 512;
    const float vv = (float)vp[(long)s * 512 + m];
#pragma unroll
    for (int j = 0; j < 16; ++j) {
      const float kk = (float)kr[d0 + j];
      acc[j] = fmaf(kk, vv, acc[j]);
      if (doks) ks[j] += kk;
    }
  }
  _Float16* outp = kvt + (long)bh * 4096;
#pragma unroll
  for (int j = 0; j < 16; ++j) outp[(long)(d0 + j) * 64 + m] = (_Float16)acc[j];
  if (doks) {
    float* ksp = ksum + (long)bh * 64 + d0;
#pragma unroll
    for (int j = 0; j < 16; ++j) ksp[j] = ks[j];
  }
}

// Z[b,l,h] = 1 / (Q[b,l,h,:] . Ksum[b,h,:] + 1e-6)
__global__ void z_kernel(const _Float16* __restrict__ q, const float* __restrict__ ksum,
                         float* __restrict__ z, int L, int total)
{
  const int idx = blockIdx.x * blockDim.x + threadIdx.x;
  if (idx >= total) return;
  const int h = idx & 7;
  const long bl = idx >> 3;
  const int b = (int)(bl / L);
  const _Float16* qp = q + bl * 512 + h * 64;
  const float* kp = ksum + (long)(b * 8 + h) * 64;
  float s = 0.f;
#pragma unroll
  for (int d = 0; d < 64; ++d) s = fmaf((float)qp[d], kp[d], s);
  z[idx] = 1.f / (s + 1e-6f);
}

// ---------------------------------------------------------------------------
// Causal attention, chunked scan. Pass A: per-chunk KV[d,m] / Ksum[d] sums.
__global__ __launch_bounds__(256) void causal_chunk_kernel(
    const _Float16* __restrict__ k, const _Float16* __restrict__ v,
    float* __restrict__ csum, int L, int nchunk, int chunk)
{
  const int gid = blockIdx.x;
  const int c = gid % nchunk;
  const int bh = gid / nchunk;
  const int b = bh >> 3, h = bh & 7;
  const long base = ((long)b * L + (long)c * chunk) * 512 + h * 64;
  const _Float16* kp = k + base;
  const _Float16* vp = v + base;
  const int d  = threadIdx.x >> 2;
  const int m0 = (threadIdx.x & 3) << 4;
  float acc[16];
#pragma unroll
  for (int j = 0; j < 16; ++j) acc[j] = 0.f;
  float ks = 0.f;
  for (int s = 0; s < chunk; ++s) {
    const float kd = (float)kp[(long)s * 512 + d];
    const _Float16* vr = vp + (long)s * 512 + m0;
#pragma unroll
    for (int j = 0; j < 16; ++j) acc[j] = fmaf(kd, (float)vr[j], acc[j]);
    ks += kd;
  }
  float* outp = csum + (long)gid * 4160 + (long)d * 64 + m0;
#pragma unroll
  for (int j = 0; j < 16; ++j) outp[j] = acc[j];
  if ((threadIdx.x & 3) == 0) csum[(long)gid * 4160 + 4096 + d] = ks;
}

// Pass B: exclusive prefix over chunks.
__global__ __launch_bounds__(256) void causal_prefix_kernel(
    const float* __restrict__ csum, float* __restrict__ cpre, int nchunk)
{
  const int bh = blockIdx.x;
  for (int idx = threadIdx.x; idx < 4160; idx += 256) {
    float run = 0.f;
    for (int c = 0; c < nchunk; ++c) {
      const long o = ((long)bh * nchunk + c) * 4160 + idx;
      cpre[o] = run;
      run += csum[o];
    }
  }
}

// Pass C: in-chunk sequential scan, 64x64 f32 state in LDS; f16 in/out.
__global__ __launch_bounds__(256) void causal_scan_kernel(
    const _Float16* __restrict__ q, const _Float16* __restrict__ k,
    const _Float16* __restrict__ v, const float* __restrict__ cpre,
    _Float16* __restrict__ o, int L, int nchunk, int chunk)
{
  __shared__ float S[64 * 64];
  __shared__ float ksum_s[64];
  __shared__ float row[192];   // q[0..63] k[64..127] v[128..191]
  const int gid = blockIdx.x;
  const int c = gid % nchunk;
  const int bh = gid / nchunk;
  const int b = bh >> 3, h = bh & 7;
  const long base = ((long)b * L + (long)c * chunk) * 512 + h * 64;
  const _Float16* qp = q + base;
  const _Float16* kp = k + base;
  const _Float16* vp = v + base;
  _Float16* op = o + base;
  const float* pre = cpre + (long)gid * 4160;
  for (int i = threadIdx.x; i < 4096; i += 256) S[i] = pre[i];
  if (threadIdx.x < 64) ksum_s[threadIdx.x] = pre[4096 + threadIdx.x];
  __syncthreads();
  const int d  = threadIdx.x >> 2;
  const int m0 = (threadIdx.x & 3) << 4;
  for (int s = 0; s < chunk; ++s) {
    const int t = threadIdx.x;
    if (t < 192) {
      const int which = t >> 6, e = t & 63;
      const _Float16* p = (which == 0) ? qp : (which == 1) ? kp : vp;
      row[t] = (float)p[(long)s * 512 + e];
    }
    __syncthreads();
    const float kd = row[64 + d];
#pragma unroll
    for (int j = 0; j < 16; ++j)
      S[d * 64 + m0 + j] = fmaf(kd, row[128 + m0 + j], S[d * 64 + m0 + j]);
    if ((threadIdx.x & 3) == 0) ksum_s[d] += kd;
    __syncthreads();
    if (t < 64) {
      float den = 1e-6f, num = 0.f;
#pragma unroll
      for (int dd = 0; dd < 64; ++dd) {
        const float qd = row[dd];
        den = fmaf(qd, ksum_s[dd], den);
        num = fmaf(qd, S[dd * 64 + t], num);
      }
      op[(long)s * 512 + t] = (_Float16)(num / den);
    }
    __syncthreads();
  }
}

// ---------------------------------------------------------------------------
// LayerNorm over 512, in-place f32, plus f16 shadow; eps=1e-5, biased var.
__global__ __launch_bounds__(256) void ln_kernel(
    float* x, _Float16* __restrict__ xh,
    const float* __restrict__ g, const float* __restrict__ bta)
{
  __shared__ float r1[256], r2[256];
  const int t = threadIdx.x;
  float* xp = x + (long)blockIdx.x * 512;
  _Float16* xhp = xh + (long)blockIdx.x * 512;
  const float a0 = xp[t], a1 = xp[t + 256];
  r1[t] = a0 + a1;
  r2[t] = a0 * a0 + a1 * a1;
  __syncthreads();
  for (int s = 128; s > 0; s >>= 1) {
    if (t < s) { r1[t] += r1[t + s]; r2[t] += r2[t + s]; }
    __syncthreads();
  }
  const float mean = r1[0] * (1.f / 512.f);
  const float var  = r2[0] * (1.f / 512.f) - mean * mean;
  const float inv  = rsqrtf(var + 1e-5f);
  const float o0 = (a0 - mean) * inv * g[t] + bta[t];
  const float o1 = (a1 - mean) * inv * g[t + 256] + bta[t + 256];
  xp[t] = o0;           xp[t + 256] = o1;
  xhp[t] = (_Float16)o0; xhp[t + 256] = (_Float16)o1;
}

// ---------------------------------------------------------------------------
namespace {

constexpr int cB = 2, cS = 2048, cT = 1024, cIN = 256, cOUT = 256;
constexpr int cH = 8, cD = 64, cDM = 512, cFF = 2048, cNL = 4;
constexpr int NCH = 8, CHK = cT / NCH;

struct Ws {
  // f32
  float *x, *y, *ksum, *zb, *csum, *cpre;
  // f16 activations
  _Float16 *xh, *yh, *srch, *tgth, *qh, *kh, *vh, *obh, *ffh, *kvth;
  // f16 transposed weights
  _Float16 *srcwT, *tgtwT, *finwT;
  _Float16 *ewT[cNL][6];    // wq,wk,wv,wo,w1,w2
  _Float16 *swT[cNL][4];    // dec self  wq,wk,wv,wo
  _Float16 *cwT[cNL][4];    // dec cross wq,wk,wv,wo
  _Float16 *dwT[cNL][2];    // dec w1,w2
};

template <int ACT, bool BIAS, bool SCALE, bool RES, bool O32, bool O16>
void gemm_launch(hipStream_t st, float* out32, _Float16* out16,
                 const _Float16* A, const _Float16* Bt,
                 const float* bias, const float* scale, const float* res,
                 int M, int N, int K, int a_rs, int out_rs,
                 long a_bo, long a_bi, long b_bo, long b_bi,
                 long o_bo, long o_bi, long sc_bo, long sc_bi, int sc_rs,
                 int batch_inner, int batches)
{
  const int tiles = (M >> 5) * (N >> 5);
  dim3 grid((unsigned)((tiles + 3) / 4), 1u, (unsigned)batches);
  gemm_wmma_kernel<ACT, BIAS, SCALE, RES, O32, O16><<<grid, 128, 0, st>>>(
      out32, out16, A, Bt, bias, scale, res, M, N, K, a_rs, out_rs,
      a_bo, a_bi, b_bo, b_bi, o_bo, o_bi, sc_bo, sc_bi, sc_rs, batch_inner);
}

template <int ACT, bool BIAS, bool RES, bool O32, bool O16>
void gemm_plain(hipStream_t st, float* out32, _Float16* out16,
                const _Float16* A, const _Float16* Bt,
                const float* bias, const float* res, int M, int N, int K)
{
  gemm_launch<ACT, BIAS, false, RES, O32, O16>(st, out32, out16, A, Bt, bias,
      nullptr, res, M, N, K, K, N, 0, 0, 0, 0, 0, 0, 0, 0, 0, 1, 1);
}

void wT(hipStream_t st, const float* src, _Float16* dst, int Nn, int Kk)
{
  dim3 grid((unsigned)(Kk / 32), (unsigned)(Nn / 32));
  w_transpose_f16_kernel<<<grid, 256, 0, st>>>(src, dst, Nn, Kk);
}

// xq32: f32 residual/out buffer; xqh: f16 A-side of same tensor (fresh)
void attn_noncausal(hipStream_t st, const Ws& w, float* xq32, const _Float16* xqh, int Lq,
                    const _Float16* xkvh, int Lkv,
                    const _Float16* wqT, const float* bq, const _Float16* wkT, const float* bk,
                    const _Float16* wvT, const float* bv, const _Float16* woT, const float* bo)
{
  gemm_plain<ACT_ELU1, true, false, false, true>(st, nullptr, w.qh, xqh,  wqT, bq, nullptr, cB * Lq,  cDM, cDM);
  gemm_plain<ACT_ELU1, true, false, false, true>(st, nullptr, w.kh, xkvh, wkT, bk, nullptr, cB * Lkv, cDM, cDM);
  gemm_plain<ACT_NONE, true, false, false, true>(st, nullptr, w.vh, xkvh, wvT, bv, nullptr, cB * Lkv, cDM, cDM);
  kvt_kernel<<<cB * cH, 256, 0, st>>>(w.kh, w.vh, w.kvth, w.ksum, Lkv);
  const int tot = cB * Lq * cH;
  z_kernel<<<(tot + 255) / 256, 256, 0, st>>>(w.qh, w.ksum, w.zb, Lq, tot);
  // O[b,l,h,m] = Z * sum_d Q[l,d]*KVT[d,m]  (batched over (b,h))
  gemm_launch<ACT_NONE, false, true, false, false, true>(st, nullptr, w.obh, w.qh, w.kvth,
      nullptr, w.zb, nullptr,
      Lq, cD, cD, cDM, cDM,
      (long)Lq * cDM, cD,
      (long)cH * cD * cD, (long)cD * cD,
      (long)Lq * cDM, cD,
      (long)Lq * cH, 1, cH,
      cH, cB * cH);
  // xq32 = xq32 + O @ wo^T + bo
  gemm_plain<ACT_NONE, true, true, true, false>(st, xq32, nullptr, w.obh, woT, bo, xq32, cB * Lq, cDM, cDM);
}

void attn_causal(hipStream_t st, const Ws& w, float* y32, const _Float16* yh,
                 const _Float16* wqT, const float* bq, const _Float16* wkT, const float* bk,
                 const _Float16* wvT, const float* bv, const _Float16* woT, const float* bo)
{
  gemm_plain<ACT_ELU1, true, false, false, true>(st, nullptr, w.qh, yh, wqT, bq, nullptr, cB * cT, cDM, cDM);
  gemm_plain<ACT_ELU1, true, false, false, true>(st, nullptr, w.kh, yh, wkT, bk, nullptr, cB * cT, cDM, cDM);
  gemm_plain<ACT_NONE, true, false, false, true>(st, nullptr, w.vh, yh, wvT, bv, nullptr, cB * cT, cDM, cDM);
  causal_chunk_kernel<<<cB * cH * NCH, 256, 0, st>>>(w.kh, w.vh, w.csum, cT, NCH, CHK);
  causal_prefix_kernel<<<cB * cH, 256, 0, st>>>(w.csum, w.cpre, NCH);
  causal_scan_kernel<<<cB * cH * NCH, 256, 0, st>>>(w.qh, w.kh, w.vh, w.cpre, w.obh, cT, NCH, CHK);
  gemm_plain<ACT_NONE, true, true, true, false>(st, y32, nullptr, w.obh, woT, bo, y32, cB * cT, cDM, cDM);
}

} // namespace

// ---------------------------------------------------------------------------
extern "C" void kernel_launch(void* const* d_in, const int* in_sizes, int n_in,
                              void* d_out, int out_size, void* d_ws, size_t ws_size,
                              hipStream_t stream)
{
  (void)in_sizes; (void)n_in; (void)out_size; (void)ws_size;

  const float* src   = (const float*)d_in[0];
  const float* tgt   = (const float*)d_in[1];
  const float* src_w = (const float*)d_in[2];
  const float* src_b = (const float*)d_in[3];
  const float* tgt_w = (const float*)d_in[4];
  const float* tgt_b = (const float*)d_in[5];
  const float* fin_w = (const float*)d_in[6];
  const float* fin_b = (const float*)d_in[7];
  // enc_params leaves (sorted pytree): attn{bk,bo,bq,bv,wk,wo,wq,wv},b1,b2,ln1{b,g},ln2{b,g},w1,w2
  const float* e_bk   = (const float*)d_in[8];
  const float* e_bo   = (const float*)d_in[9];
  const float* e_bq   = (const float*)d_in[10];
  const float* e_bv   = (const float*)d_in[11];
  const float* e_wk   = (const float*)d_in[12];
  const float* e_wo   = (const float*)d_in[13];
  const float* e_wq   = (const float*)d_in[14];
  const float* e_wv   = (const float*)d_in[15];
  const float* e_b1   = (const float*)d_in[16];
  const float* e_b2   = (const float*)d_in[17];
  const float* e_ln1b = (const float*)d_in[18];
  const float* e_ln1g = (const float*)d_in[19];
  const float* e_ln2b = (const float*)d_in[20];
  const float* e_ln2g = (const float*)d_in[21];
  const float* e_w1   = (const float*)d_in[22];
  const float* e_w2   = (const float*)d_in[23];
  // dec_params leaves: b1,b2,cross{...},ln1,ln2,ln3,self{...},w1,w2
  const float* dd_b1  = (const float*)d_in[24];
  const float* dd_b2  = (const float*)d_in[25];
  const float* c_bk   = (const float*)d_in[26];
  const float* c_bo   = (const float*)d_in[27];
  const float* c_bq   = (const float*)d_in[28];
  const float* c_bv   = (const float*)d_in[29];
  const float* c_wk   = (const float*)d_in[30];
  const float* c_wo   = (const float*)d_in[31];
  const float* c_wq   = (const float*)d_in[32];
  const float* c_wv   = (const float*)d_in[33];
  const float* d_ln1b = (const float*)d_in[34];
  const float* d_ln1g = (const float*)d_in[35];
  const float* d_ln2b = (const float*)d_in[36];
  const float* d_ln2g = (const float*)d_in[37];
  const float* d_ln3b = (const float*)d_in[38];
  const float* d_ln3g = (const float*)d_in[39];
  const float* s_bk   = (const float*)d_in[40];
  const float* s_bo   = (const float*)d_in[41];
  const float* s_bq   = (const float*)d_in[42];
  const float* s_bv   = (const float*)d_in[43];
  const float* s_wk   = (const float*)d_in[44];
  const float* s_wo   = (const float*)d_in[45];
  const float* s_wq   = (const float*)d_in[46];
  const float* s_wv   = (const float*)d_in[47];
  const float* dd_w1  = (const float*)d_in[48];
  const float* dd_w2  = (const float*)d_in[49];

  Ws w;
  // ---- f32 region ----
  float* pf = (float*)d_ws;
  auto takef = [&](size_t n) { float* r = pf; pf += n; return r; };
  w.x    = takef((size_t)cB * cS * cDM);
  w.y    = takef((size_t)cB * cT * cDM);
  w.ksum = takef((size_t)cB * cH * cD);
  w.zb   = takef((size_t)cB * cS * cH);
  w.csum = takef((size_t)cB * cH * NCH * 4160);
  w.cpre = takef((size_t)cB * cH * NCH * 4160);
  // ---- f16 region (16B aligned) ----
  uintptr_t up = ((uintptr_t)pf + 15) & ~(uintptr_t)15;
  _Float16* ph = (_Float16*)up;
  auto takeh = [&](size_t n) { _Float16* r = ph; ph += n; return r; };
  w.xh   = takeh((size_t)cB * cS * cDM);
  w.yh   = takeh((size_t)cB * cT * cDM);
  w.srch = takeh((size_t)cB * cS * cIN);
  w.tgth = takeh((size_t)cB * cT * cOUT);
  w.qh   = takeh((size_t)cB * cS * cDM);
  w.kh   = takeh((size_t)cB * cS * cDM);
  w.vh   = takeh((size_t)cB * cS * cDM);
  w.obh  = takeh((size_t)cB * cS * cDM);
  w.ffh  = takeh((size_t)cB * cS * cFF);
  w.kvth = takeh((size_t)cB * cH * cD * cD);
  w.srcwT = takeh((size_t)cIN * cDM);
  w.tgtwT = takeh((size_t)cOUT * cDM);
  w.finwT = takeh((size_t)cDM * cOUT);
  for (int i = 0; i < cNL; ++i) {
    for (int j = 0; j < 4; ++j) w.ewT[i][j] = takeh((size_t)cDM * cDM);
    w.ewT[i][4] = takeh((size_t)cDM * cFF);
    w.ewT[i][5] = takeh((size_t)cFF * cDM);
    for (int j = 0; j < 4; ++j) w.swT[i][j] = takeh((size_t)cDM * cDM);
    for (int j = 0; j < 4; ++j) w.cwT[i][j] = takeh((size_t)cDM * cDM);
    w.dwT[i][0] = takeh((size_t)cDM * cFF);
    w.dwT[i][1] = takeh((size_t)cFF * cDM);
  }

  // ---- one-time (per launch) weight transpose-convert to f16 [K,N] ----
  wT(stream, src_w, w.srcwT, cDM, cIN);
  wT(stream, tgt_w, w.tgtwT, cDM, cOUT);
  wT(stream, fin_w, w.finwT, cOUT, cDM);
  for (int i = 0; i < cNL; ++i) {
    const long wo = (long)i * cDM * cDM, fo = (long)i * cDM * cFF;
    wT(stream, e_wq + wo, w.ewT[i][0], cDM, cDM);
    wT(stream, e_wk + wo, w.ewT[i][1], cDM, cDM);
    wT(stream, e_wv + wo, w.ewT[i][2], cDM, cDM);
    wT(stream, e_wo + wo, w.ewT[i][3], cDM, cDM);
    wT(stream, e_w1 + fo, w.ewT[i][4], cFF, cDM);
    wT(stream, e_w2 + fo, w.ewT[i][5], cDM, cFF);
    wT(stream, s_wq + wo, w.swT[i][0], cDM, cDM);
    wT(stream, s_wk + wo, w.swT[i][1], cDM, cDM);
    wT(stream, s_wv + wo, w.swT[i][2], cDM, cDM);
    wT(stream, s_wo + wo, w.swT[i][3], cDM, cDM);
    wT(stream, c_wq + wo, w.cwT[i][0], cDM, cDM);
    wT(stream, c_wk + wo, w.cwT[i][1], cDM, cDM);
    wT(stream, c_wv + wo, w.cwT[i][2], cDM, cDM);
    wT(stream, c_wo + wo, w.cwT[i][3], cDM, cDM);
    wT(stream, dd_w1 + fo, w.dwT[i][0], cFF, cDM);
    wT(stream, dd_w2 + fo, w.dwT[i][1], cDM, cFF);
  }
  {
    const long n1 = (long)cB * cS * cIN;
    cvt_f16_kernel<<<(unsigned)((n1 + 255) / 256), 256, 0, stream>>>(src, w.srch, n1);
    const long n2 = (long)cB * cT * cOUT;
    cvt_f16_kernel<<<(unsigned)((n2 + 255) / 256), 256, 0, stream>>>(tgt, w.tgth, n2);
  }

  // x = gelu(src @ src_w^T + src_b)   (f32 + f16 shadow)
  gemm_plain<ACT_GELU, true, false, true, true>(stream, w.x, w.xh, w.srch, w.srcwT,
                                                src_b, nullptr, cB * cS, cDM, cIN);

  // Encoder stack
  for (int i = 0; i < cNL; ++i) {
    const long bOff = (long)i * cDM;
    attn_noncausal(stream, w, w.x, w.xh, cS, w.xh, cS,
                   w.ewT[i][0], e_bq + bOff, w.ewT[i][1], e_bk + bOff,
                   w.ewT[i][2], e_bv + bOff, w.ewT[i][3], e_bo + bOff);
    ln_kernel<<<cB * cS, 256, 0, stream>>>(w.x, w.xh, e_ln1g + bOff, e_ln1b + bOff);
    gemm_plain<ACT_RELU, true, false, false, true>(stream, nullptr, w.ffh, w.xh, w.ewT[i][4],
                                                   e_b1 + (long)i * cFF, nullptr, cB * cS, cFF, cDM);
    gemm_plain<ACT_NONE, true, true, true, false>(stream, w.x, nullptr, w.ffh, w.ewT[i][5],
                                                  e_b2 + bOff, w.x, cB * cS, cDM, cFF);
    ln_kernel<<<cB * cS, 256, 0, stream>>>(w.x, w.xh, e_ln2g + bOff, e_ln2b + bOff);
  }

  // y = gelu(tgt @ tgt_w^T + tgt_b)
  gemm_plain<ACT_GELU, true, false, true, true>(stream, w.y, w.yh, w.tgth, w.tgtwT,
                                                tgt_b, nullptr, cB * cT, cDM, cOUT);

  // Decoder stack (mem = x / xh, frozen after encoder)
  for (int i = 0; i < cNL; ++i) {
    const long bOff = (long)i * cDM;
    attn_causal(stream, w, w.y, w.yh,
                w.swT[i][0], s_bq + bOff, w.swT[i][1], s_bk + bOff,
                w.swT[i][2], s_bv + bOff, w.swT[i][3], s_bo + bOff);
    ln_kernel<<<cB * cT, 256, 0, stream>>>(w.y, w.yh, d_ln1g + bOff, d_ln1b + bOff);
    attn_noncausal(stream, w, w.y, w.yh, cT, w.xh, cS,
                   w.cwT[i][0], c_bq + bOff, w.cwT[i][1], c_bk + bOff,
                   w.cwT[i][2], c_bv + bOff, w.cwT[i][3], c_bo + bOff);
    ln_kernel<<<cB * cT, 256, 0, stream>>>(w.y, w.yh, d_ln2g + bOff, d_ln2b + bOff);
    gemm_plain<ACT_RELU, true, false, false, true>(stream, nullptr, w.ffh, w.yh, w.dwT[i][0],
                                                   dd_b1 + (long)i * cFF, nullptr, cB * cT, cFF, cDM);
    gemm_plain<ACT_NONE, true, true, true, false>(stream, w.y, nullptr, w.ffh, w.dwT[i][1],
                                                  dd_b2 + bOff, w.y, cB * cT, cDM, cFF);
    ln_kernel<<<cB * cT, 256, 0, stream>>>(w.y, w.yh, d_ln3g + bOff, d_ln3b + bOff);
  }

  // out = gelu(y @ fin_w^T + fin_b)  (f32 only)
  gemm_plain<ACT_GELU, true, false, true, false>(stream, (float*)d_out, nullptr, w.yh, w.finwT,
                                                 fin_b, nullptr, cB * cT, cOUT, cDM);
}
